// ManifoldAttentionLayer_54039278518589
// MI455X (gfx1250) — compile-verified, matrix operand order checked by reading, not run on hardware
//
#include <hip/hip_runtime.h>
#include <hip/hip_bf16.h>
#include <stdint.h>

// ---------------------------------------------------------------------------
// ManifoldAttentionLayer for MI455X (gfx1250, wave32, WMMA bf16)
// B=4, L=1024, D=1024, H=16, hd=64, M=64 (gram), K=32 (harmonics)
//  - all matmuls on v_wmma_f32_16x16x32_bf16 (f32 accumulate)
//  - flash-attention fusion (no 256MB score tensor)
//  - LDS tiles staged via global_load_async_to_lds_b128 (ASYNCcnt),
//    double-buffered; falls back to plain LDS staging if the async
//    builtins are unavailable on this toolchain.
// ---------------------------------------------------------------------------

typedef __attribute__((ext_vector_type(16))) __bf16 v16bf;
typedef __attribute__((ext_vector_type(8)))  float  v8f;
typedef __attribute__((ext_vector_type(4)))  int    v4i;

#define D_MODEL   1024
#define N_HEADSC  16
#define HEAD_DIM  64
#define SEQ       1024
#define BATCH     4
#define BIAS_SCL  0.1f

#if defined(__AMDGCN__) && \
    __has_builtin(__builtin_amdgcn_global_load_async_to_lds_b128) && \
    __has_builtin(__builtin_amdgcn_s_wait_asynccnt)
#define ASYNC_LDS 1
typedef __attribute__((address_space(1))) v4i gv4i;   // global 16B chunk
typedef __attribute__((address_space(3))) v4i lv4i;   // LDS 16B chunk
#else
#define ASYNC_LDS 0
#endif

__device__ __forceinline__ void async_copy_b128(const unsigned short* g,
                                                unsigned short* l) {
#if ASYNC_LDS
  // generic->AS1 via 64-bit integer; generic LDS pointer's low 32 bits are the
  // LDS offset, so the AS3 pointer is the truncated integer value.
  __builtin_amdgcn_global_load_async_to_lds_b128(
      (gv4i*)(uintptr_t)g,
      (lv4i*)(unsigned int)(uintptr_t)l,
      0, 0);
#else
  *(uint4*)l = *(const uint4*)g;
#endif
}

__device__ __forceinline__ void wait_async_copies() {
#if ASYNC_LDS
  __builtin_amdgcn_s_wait_asynccnt(0);
#endif
}

__device__ __forceinline__ unsigned short f32_to_bf16(float f) {
  unsigned int u = __float_as_uint(f);
  unsigned int r = ((u >> 16) & 1u) + 0x7FFFu;   // round-to-nearest-even
  return (unsigned short)((u + r) >> 16);
}

// ---------------------------------------------------------------------------
__global__ void cvt_bf16_kernel(const float* __restrict__ in,
                                unsigned short* __restrict__ out, int n) {
  int i = blockIdx.x * blockDim.x + threadIdx.x;
  if (i < n) out[i] = f32_to_bf16(in[i]);
}

// ---------------------------------------------------------------------------
// gram[b,m,n] = sum_k harm[b,m,k]*harm[b,n,k]   (64x64x32 per batch, tiny)
__global__ void gram_kernel(const float* __restrict__ harm,
                            float* __restrict__ gram) {
  __shared__ float sh[64 * 32];
  int b = blockIdx.x;
  for (int i = threadIdx.x; i < 2048; i += 256) sh[i] = harm[b * 2048 + i];
  __syncthreads();
  for (int idx = threadIdx.x; idx < 4096; idx += 256) {
    int m = idx >> 6, n = idx & 63;
    float s = 0.f;
#pragma unroll
    for (int k2 = 0; k2 < 32; ++k2) s += sh[m * 32 + k2] * sh[n * 32 + k2];
    gram[b * 4096 + idx] = s;
  }
}

// ---------------------------------------------------------------------------
// C[M,N] = A[M,K] * W[N,K]^T + bias   (M=4096, N=K=1024, bf16 in, f32 acc)
// Weight tile (64 N-rows x 64 K) staged in LDS via async copy, double
// buffered, shared by all 8 waves of the block (8x reuse).
// mode 0: bf16 out in [B,H,L,hd] layout (Q,K)
// mode 1: f32  out row-major [M,N]      (final projection)
// mode 2: bf16 out in [B,H,hd,L] layout (V transposed for attention)
__global__ void __launch_bounds__(256)
gemm_bf16_kernel(const unsigned short* __restrict__ A,
                 const unsigned short* __restrict__ W,
                 const float* __restrict__ bias,
                 unsigned short* __restrict__ outb,
                 float* __restrict__ outf, int mode) {
  __shared__ __align__(16) unsigned short s_w[2][64 * 64];  // 16 KB

  const int tid = threadIdx.x;
  const int w = tid >> 5, lane = tid & 31, l16 = lane & 15, lh = lane >> 4;
  const int n0 = blockIdx.x * 64;
  const int m0 = blockIdx.y * 128 + w * 16;

  // stage first weight tile
  for (int i = tid; i < 512; i += 256) {
    int row = i >> 3, c = i & 7;
    async_copy_b128(W + (size_t)(n0 + row) * D_MODEL + c * 8,
                    &s_w[0][row * 64 + c * 8]);
  }
  wait_async_copies();
  __syncthreads();

  v8f acc[4] = {};
  const unsigned short* arow = A + (size_t)(m0 + l16) * D_MODEL + lh * 16;

  for (int kt = 0; kt < D_MODEL / 64; ++kt) {
    const int cur = kt & 1;
    if (kt + 1 < D_MODEL / 64) {
      const int kk = (kt + 1) * 64;
      for (int i = tid; i < 512; i += 256) {
        int row = i >> 3, c = i & 7;
        async_copy_b128(W + (size_t)(n0 + row) * D_MODEL + kk + c * 8,
                        &s_w[cur ^ 1][row * 64 + c * 8]);
      }
    }
#pragma unroll
    for (int ks = 0; ks < 2; ++ks) {
      v16bf af = *(const v16bf*)(arow + kt * 64 + ks * 32);
#pragma unroll
      for (int nt = 0; nt < 4; ++nt) {
        v16bf bf = *(const v16bf*)(&s_w[cur][(nt * 16 + l16) * 64 + ks * 32 + lh * 16]);
        acc[nt] = __builtin_amdgcn_wmma_f32_16x16x32_bf16(
            false, af, false, bf, (short)0, acc[nt], false, false);
      }
    }
    wait_async_copies();
    __syncthreads();
  }

#pragma unroll
  for (int nt = 0; nt < 4; ++nt) {
    int col = n0 + nt * 16 + l16;
    float bv = bias[col];
#pragma unroll
    for (int r = 0; r < 8; ++r) {
      float val = acc[nt][r] + bv;
      int row = m0 + r + 8 * lh;
      int b = row >> 10, l = row & 1023, h = col >> 6, e = col & 63;
      if (mode == 0) {
        outb[((size_t)(b * N_HEADSC + h) * SEQ + l) * HEAD_DIM + e] =
            f32_to_bf16(val);
      } else if (mode == 2) {
        outb[((size_t)(b * N_HEADSC + h) * HEAD_DIM + e) * SEQ + l] =
            f32_to_bf16(val);
      } else {
        outf[(size_t)row * D_MODEL + col] = val;
      }
    }
  }
}

// ---------------------------------------------------------------------------
// Fused flash attention with bilinear gram bias.
// grid = (L/64, B*H), block = 128 (4 waves). Each wave owns 16 query rows.
// K tile [key][hd] and Vt tile [hd][key] double-buffered in LDS via async
// copies; next KV tile is in flight while current tile is computed.
__global__ void __launch_bounds__(128)
attn_kernel(const unsigned short* __restrict__ Q,
            const unsigned short* __restrict__ K,
            const unsigned short* __restrict__ Vt,
            const float* __restrict__ gram,
            unsigned short* __restrict__ attn_out) {
  __shared__ __align__(16) float          s_gram[64 * 64];     // 16 KB
  __shared__ __align__(16) unsigned short s_k[2][64 * 64];     // 16 KB
  __shared__ __align__(16) unsigned short s_vt[2][64 * 64];    // 16 KB
  __shared__ __align__(16) unsigned short s_p[4][16 * 64];     // 8 KB

  const int tid = threadIdx.x;
  const int w = tid >> 5, lane = tid & 31, l16 = lane & 15, lh = lane >> 4;
  const int qt = blockIdx.x;
  const int bh = blockIdx.y;
  const int b = bh >> 4, h = bh & 15;
  const int m0 = qt * 64 + w * 16;

  const unsigned short* gq  = Q  + (size_t)bh * SEQ * HEAD_DIM;
  const unsigned short* gk  = K  + (size_t)bh * SEQ * HEAD_DIM;
  const unsigned short* gvt = Vt + (size_t)bh * HEAD_DIM * SEQ;

  auto stage = [&](int buf, int kt2) {
    // K tile: 64 keys x 64 hd, fully contiguous -> 512 b128 chunks
    for (int i = tid; i < 512; i += 128)
      async_copy_b128(gk + (size_t)kt2 * 4096 + i * 8, &s_k[buf][i * 8]);
    // Vt tile: 64 hd-rows x 64 keys, row stride SEQ
    for (int i = tid; i < 512; i += 128) {
      int e = i >> 3, c = i & 7;
      async_copy_b128(gvt + (size_t)e * SEQ + kt2 * 64 + c * 8,
                      &s_vt[buf][e * 64 + c * 8]);
    }
  };

  for (int i = tid; i < 4096; i += 128) s_gram[i] = gram[b * 4096 + i];
  stage(0, 0);

  v16bf qa[2];
  {
    const unsigned short* qrow = gq + (size_t)(m0 + l16) * HEAD_DIM + lh * 16;
    qa[0] = *(const v16bf*)(qrow);
    qa[1] = *(const v16bf*)(qrow + 32);
  }

  // bilinear row coefficients (1024 -> 64, align_corners=False)
  int ry0[8], ry1[8];
  float wy[8];
#pragma unroll
  for (int r = 0; r < 8; ++r) {
    int R = m0 + r + 8 * lh;
    float sy = (R + 0.5f) * 0.0625f - 0.5f;
    sy = fmaxf(sy, 0.f);
    int i0 = (int)sy; if (i0 > 63) i0 = 63;
    int i1 = i0 + 1;  if (i1 > 63) i1 = 63;
    ry0[r] = i0; ry1[r] = i1; wy[r] = sy - (float)i0;
  }

  float rowmax[8], rowsum[8];
#pragma unroll
  for (int r = 0; r < 8; ++r) { rowmax[r] = -3.0e38f; rowsum[r] = 0.f; }
  v8f o[4] = {};

  wait_async_copies();
  __syncthreads();

  for (int kt = 0; kt < SEQ / 64; ++kt) {
    const int cur = kt & 1;
    if (kt + 1 < SEQ / 64) stage(cur ^ 1, kt + 1);

    // S = Q * K^T  (16x64 per wave)
    v8f sc[4] = {};
#pragma unroll
    for (int ks = 0; ks < 2; ++ks) {
#pragma unroll
      for (int nt = 0; nt < 4; ++nt) {
        v16bf kb = *(const v16bf*)(&s_k[cur][(nt * 16 + l16) * HEAD_DIM + ks * 32 + lh * 16]);
        sc[nt] = __builtin_amdgcn_wmma_f32_16x16x32_bf16(
            false, qa[ks], false, kb, (short)0, sc[nt], false, false);
      }
    }

    // scale + bilinear gram bias
#pragma unroll
    for (int nt = 0; nt < 4; ++nt) {
      int C = kt * 64 + nt * 16 + l16;
      float sx = (C + 0.5f) * 0.0625f - 0.5f;
      sx = fmaxf(sx, 0.f);
      int c0 = (int)sx; if (c0 > 63) c0 = 63;
      int c1 = c0 + 1;  if (c1 > 63) c1 = 63;
      float wx = sx - (float)c0;
#pragma unroll
      for (int r = 0; r < 8; ++r) {
        float g00 = s_gram[ry0[r] * 64 + c0], g01 = s_gram[ry0[r] * 64 + c1];
        float g10 = s_gram[ry1[r] * 64 + c0], g11 = s_gram[ry1[r] * 64 + c1];
        float top = g00 + (g01 - g00) * wx;
        float bot = g10 + (g11 - g10) * wx;
        float bias = top + (bot - top) * wy[r];
        sc[nt][r] = sc[nt][r] * 0.125f + BIAS_SCL * bias;
      }
    }

    // online softmax (reduce across 16-lane N dimension of C layout)
#pragma unroll
    for (int r = 0; r < 8; ++r) {
      float vmax = fmaxf(fmaxf(sc[0][r], sc[1][r]), fmaxf(sc[2][r], sc[3][r]));
#pragma unroll
      for (int m = 1; m < 16; m <<= 1) vmax = fmaxf(vmax, __shfl_xor(vmax, m, 32));
      float mnew = fmaxf(rowmax[r], vmax);
      float corr = __expf(rowmax[r] - mnew);
      rowmax[r] = mnew;
      float p0 = __expf(sc[0][r] - mnew);
      float p1 = __expf(sc[1][r] - mnew);
      float p2 = __expf(sc[2][r] - mnew);
      float p3 = __expf(sc[3][r] - mnew);
      float ls = (p0 + p1) + (p2 + p3);
#pragma unroll
      for (int m = 1; m < 16; m <<= 1) ls += __shfl_xor(ls, m, 32);
      rowsum[r] = rowsum[r] * corr + ls;
      o[0][r] *= corr; o[1][r] *= corr; o[2][r] *= corr; o[3][r] *= corr;
      int rl = r + 8 * lh;
      s_p[w][rl * 64 +  0 + l16] = f32_to_bf16(p0);
      s_p[w][rl * 64 + 16 + l16] = f32_to_bf16(p1);
      s_p[w][rl * 64 + 32 + l16] = f32_to_bf16(p2);
      s_p[w][rl * 64 + 48 + l16] = f32_to_bf16(p3);
    }

    // O += P * V   (per-wave LDS P; in-wave DS ordering keeps RAW safe)
#pragma unroll
    for (int ks = 0; ks < 2; ++ks) {
      v16bf pa = *(const v16bf*)(&s_p[w][l16 * 64 + ks * 32 + lh * 16]);
#pragma unroll
      for (int nt = 0; nt < 4; ++nt) {
        v16bf vbf = *(const v16bf*)(&s_vt[cur][(nt * 16 + l16) * 64 + ks * 32 + lh * 16]);
        o[nt] = __builtin_amdgcn_wmma_f32_16x16x32_bf16(
            false, pa, false, vbf, (short)0, o[nt], false, false);
      }
    }

    wait_async_copies();
    __syncthreads();
  }

  // normalize and write [B, L, D] bf16 for the output projection
#pragma unroll
  for (int r = 0; r < 8; ++r) {
    float inv = 1.0f / rowsum[r];
    int l = m0 + r + 8 * lh;
#pragma unroll
    for (int nt = 0; nt < 4; ++nt) {
      int col = h * HEAD_DIM + nt * 16 + l16;
      attn_out[((size_t)(b * SEQ + l)) * D_MODEL + col] =
          f32_to_bf16(o[nt][r] * inv);
    }
  }
}

// ---------------------------------------------------------------------------
extern "C" void kernel_launch(void* const* d_in, const int* in_sizes, int n_in,
                              void* d_out, int out_size, void* d_ws, size_t ws_size,
                              hipStream_t stream) {
  (void)in_sizes; (void)n_in; (void)out_size; (void)ws_size;
  const float* x    = (const float*)d_in[0];
  const float* harm = (const float*)d_in[1];
  const float* wq   = (const float*)d_in[2];
  const float* bq   = (const float*)d_in[3];
  const float* wk   = (const float*)d_in[4];
  const float* bk   = (const float*)d_in[5];
  const float* wv   = (const float*)d_in[6];
  const float* bv   = (const float*)d_in[7];
  const float* wo   = (const float*)d_in[8];
  const float* bo   = (const float*)d_in[9];
  float* out = (float*)d_out;

  char* ws = (char*)d_ws;
  unsigned short* xb   = (unsigned short*)(ws + (size_t)0);
  unsigned short* wqb  = (unsigned short*)(ws + ((size_t)8  << 20));
  unsigned short* wkb  = (unsigned short*)(ws + ((size_t)10 << 20));
  unsigned short* wvb  = (unsigned short*)(ws + ((size_t)12 << 20));
  unsigned short* wob  = (unsigned short*)(ws + ((size_t)14 << 20));
  unsigned short* qb   = (unsigned short*)(ws + ((size_t)16 << 20));
  unsigned short* kb   = (unsigned short*)(ws + ((size_t)24 << 20));
  unsigned short* vtb  = (unsigned short*)(ws + ((size_t)32 << 20));
  unsigned short* ab   = (unsigned short*)(ws + ((size_t)40 << 20));
  float*          gram = (float*)        (ws + ((size_t)48 << 20));

  const int nX = BATCH * SEQ * D_MODEL;       // 4194304
  const int nW = D_MODEL * D_MODEL;           // 1048576
  cvt_bf16_kernel<<<(nX + 255) / 256, 256, 0, stream>>>(x,  xb,  nX);
  cvt_bf16_kernel<<<(nW + 255) / 256, 256, 0, stream>>>(wq, wqb, nW);
  cvt_bf16_kernel<<<(nW + 255) / 256, 256, 0, stream>>>(wk, wkb, nW);
  cvt_bf16_kernel<<<(nW + 255) / 256, 256, 0, stream>>>(wv, wvb, nW);
  cvt_bf16_kernel<<<(nW + 255) / 256, 256, 0, stream>>>(wo, wob, nW);

  gram_kernel<<<BATCH, 256, 0, stream>>>(harm, gram);

  dim3 gg(D_MODEL / 64, (BATCH * SEQ) / 128);  // (16, 32)
  gemm_bf16_kernel<<<gg, 256, 0, stream>>>(xb, wqb, bq, qb,  nullptr, 0);
  gemm_bf16_kernel<<<gg, 256, 0, stream>>>(xb, wkb, bk, kb,  nullptr, 0);
  gemm_bf16_kernel<<<gg, 256, 0, stream>>>(xb, wvb, bv, vtb, nullptr, 2);

  attn_kernel<<<dim3(SEQ / 64, BATCH * N_HEADSC), 128, 0, stream>>>(
      qb, kb, vtb, gram, ab);

  gemm_bf16_kernel<<<gg, 256, 0, stream>>>(ab, wob, bo, nullptr, out, 1);
}
